// Qwen3AttentionModified_75093208203860
// MI455X (gfx1250) — matrix operand order repaired and checked by reading, not compiled, not run on hardware
//
#include <hip/hip_runtime.h>

// ---------------------------------------------------------------------------
// CDNA5 / gfx1250 wave32 WMMA attention pipeline
//   - one-time fp32->bf16 conversion of all matmul operands
//   - LDS tile staging via global_load_async_to_lds_b128 (ASYNCcnt),
//     double-buffered and overlapped with WMMA via partial asynccnt waits
//   - bf16 WMMA, f32 accumulate, fused RoPE / pow8 epilogues in registers
// ---------------------------------------------------------------------------

typedef __attribute__((ext_vector_type(16))) __bf16 v16bf;
typedef __attribute__((ext_vector_type(8)))  float  v8f;

union Frag {
    v16bf v;
    unsigned int u[8];
};

__device__ __forceinline__ unsigned short f2bf(float f) {
    unsigned int u = __float_as_uint(f);
    u += 0x7FFFu + ((u >> 16) & 1u);      // round-to-nearest-even
    return (unsigned short)(u >> 16);
}

// LDS byte offset of a __shared__ object: flat shared addresses carry the
// wave-relative LDS offset in the low 32 bits (aperture bits are in [63:32]).
__device__ __forceinline__ unsigned lds_off(const void* p) {
    return (unsigned)(unsigned long long)p;
}

// Async 16-byte copy global -> LDS (GLOBAL_LOAD_ASYNC_TO_LDS_B128, ASYNCcnt).
__device__ __forceinline__ void async_ld16(unsigned ldsByteOff, const void* gaddr) {
    asm volatile("global_load_async_to_lds_b128 %0, %1, off"
                 :: "v"(ldsByteOff), "v"(gaddr)
                 : "memory");
}

// Partial waits: async loads retire in order, so waiting "<= N" guarantees
// everything older than the newest N transfers has landed in LDS.
__device__ __forceinline__ void wait_async_le0() {
    asm volatile("s_wait_asynccnt 0x0" ::: "memory");
}
__device__ __forceinline__ void wait_async_le3() {
    asm volatile("s_wait_asynccnt 0x3" ::: "memory");
}
__device__ __forceinline__ void wait_async_le8() {
    asm volatile("s_wait_asynccnt 0x8" ::: "memory");
}

// A operand: 16x32 bf16. Lane l (m = l&15, lh = l>>4), VGPR j holds bf16 pair:
//   j<4 : K = kBase + 2j + 8*lh ; j>=4 : K = kBase + 16 + 2(j-4) + 8*lh
__device__ __forceinline__ void load_a_frag(Frag& a, const unsigned short* lds,
                                            int rowBase, int strideShorts,
                                            int kBase, int lane) {
    const int row = rowBase + (lane & 15);
    const int lh  = (lane >> 4) & 1;
    const unsigned int* rp =
        (const unsigned int*)(lds + (size_t)row * strideShorts) + (kBase >> 1);
#pragma unroll
    for (int j = 0; j < 4; ++j) {
        a.u[j]     = rp[j + 4 * lh];
        a.u[4 + j] = rp[8 + j + 4 * lh];
    }
}

// B operand: 32x16 bf16. Lane l: N = l&15, lh = l>>4; VGPR j holds K pair
//   j<4 : K = 2j + 8*lh ; j>=4 : K = 16 + 2(j-4) + 8*lh
__device__ __forceinline__ void load_b_frag(Frag& b, const unsigned short* lds,
                                            int kRowBase, int strideShorts,
                                            int nBase, int lane) {
    const int n  = nBase + (lane & 15);
    const int lh = (lane >> 4) & 1;
#pragma unroll
    for (int j = 0; j < 8; ++j) {
        const int k0 = kRowBase + (j < 4 ? 2 * j : 16 + 2 * (j - 4)) + 8 * lh;
        b.u[j] = (unsigned int)lds[(size_t)k0 * strideShorts + n] |
                 ((unsigned int)lds[(size_t)(k0 + 1) * strideShorts + n] << 16);
    }
}

// ---------------------------------------------------------------------------
// fp32 -> bf16 one-time conversion (vectorized by 4)
// ---------------------------------------------------------------------------
__global__ __launch_bounds__(256) void f32_to_bf16(const float* __restrict__ s,
                                                   unsigned short* __restrict__ d,
                                                   int n) {
    const int i = (blockIdx.x * 256 + threadIdx.x) * 4;
    if (i < n) {
        const float4 v = *(const float4*)(s + i);
        d[i + 0] = f2bf(v.x);
        d[i + 1] = f2bf(v.y);
        d[i + 2] = f2bf(v.z);
        d[i + 3] = f2bf(v.w);
    }
}

// ---------------------------------------------------------------------------
// Generic 128x128-tile bf16 WMMA GEMM, double-buffered async-LDS staging.
// mode 0: plain fp32 store  outf[s*N + n]
// mode 1: Q: *denom*norm, RoPE, bf16 store Q[h][s][128]
// mode 2: K: *denom*norm, RoPE, bf16 store K^T[h][d][S]
// mode 3: V: bf16 store V[h][s][128]
// ---------------------------------------------------------------------------
#define A_STRIDE 40     // 32 + 8 pad shorts  (80 B rows, 16B-aligned)
#define B_STRIDE 136    // 128 + 8 pad shorts (272 B rows)

__global__ __launch_bounds__(256) void gemm_bf16_wmma(
    const unsigned short* __restrict__ A, const unsigned short* __restrict__ B,
    float* __restrict__ outf, unsigned short* __restrict__ outb,
    int M, int K, int N, int mode,
    const float* __restrict__ cosp, const float* __restrict__ sinp,
    const float* __restrict__ normw, const float* __restrict__ denom,
    int S, int denomStride)
{
    __shared__ __align__(16) unsigned short Asub[2][128 * A_STRIDE]; // 128 x 32
    __shared__ __align__(16) unsigned short Bsub[2][32 * B_STRIDE];  // 32 x 128

    const int tid  = threadIdx.x;
    const int lane = tid & 31;
    const int wave = tid >> 5;             // 0..7, each owns 16 rows
    const int n0 = blockIdx.x * 128;
    const int m0 = blockIdx.y * 128;

    const unsigned aBase0 = lds_off(Asub[0]), aBase1 = lds_off(Asub[1]);
    const unsigned bBase0 = lds_off(Bsub[0]), bBase1 = lds_off(Bsub[1]);

    v8f acc[8];
#pragma unroll
    for (int t = 0; t < 8; ++t)
#pragma unroll
        for (int v = 0; v < 8; ++v) acc[t][v] = 0.0f;

    // staging geometry (256 threads): 3 x b128 per thread per tile
    const int arow = tid >> 1, aseg = (tid & 1) * 2;
    const int brow = tid >> 4, bseg = tid & 15;
    const unsigned short* agp0 = A + (size_t)(m0 + arow) * K;
    const unsigned short* bgp0 = B + (size_t)brow * N + n0;

    // stage one 32-K tile into buffer `buf`
    auto stage = [&](int buf, int kb) {
        const unsigned aB = buf ? aBase1 : aBase0;
        const unsigned bB = buf ? bBase1 : bBase0;
#pragma unroll
        for (int j = 0; j < 2; ++j)
            async_ld16(aB + arow * (A_STRIDE * 2) + (aseg + j) * 16,
                       agp0 + kb + (aseg + j) * 8);
        async_ld16(bB + brow * (B_STRIDE * 2) + bseg * 16,
                   bgp0 + (size_t)kb * N + bseg * 8);
    };

    const int nk = K >> 5;
    stage(0, 0);

    for (int i = 0; i < nk; ++i) {
        if (i + 1 < nk) {
            stage((i + 1) & 1, (i + 1) << 5);   // overlap next tile's DMA
            if ((i + 2) << 5 < K) __builtin_prefetch(agp0 + ((i + 2) << 5), 0, 1);
            wait_async_le3();                   // only tile i must have landed
        } else {
            wait_async_le0();
        }
        __syncthreads();

        const unsigned short* As = Asub[i & 1];
        const unsigned short* Bs = Bsub[i & 1];
        Frag a, b;
        load_a_frag(a, As, wave * 16, A_STRIDE, 0, lane);
#pragma unroll
        for (int t = 0; t < 8; ++t) {
            load_b_frag(b, Bs, 0, B_STRIDE, t * 16, lane);
            acc[t] = __builtin_amdgcn_wmma_f32_16x16x32_bf16(
                false, a.v, false, b.v, (short)0, acc[t], false, false);
        }
        __syncthreads();
    }

    // D layout: element (m, n): lane = n + 16*(m/8), vgpr = m%8
    const int mloc = wave * 16 + ((lane >> 4) & 1) * 8;
    const int nloc = lane & 15;

    if (mode == 1 || mode == 2) {
        const int h = n0 >> 7;                 // 128-wide tile == one head
#pragma unroll
        for (int t = 0; t < 8; ++t) {
            const int d = t * 16 + nloc;
            const float nw = normw[d];
#pragma unroll
            for (int v = 0; v < 8; ++v) {
                const int s = m0 + mloc + v;
                acc[t][v] *= denom[(size_t)s * denomStride + h] * nw;
            }
        }
        // RoPE: d and d+64 live in acc[t] / acc[t+4] at same lane+vgpr.
#pragma unroll
        for (int t = 0; t < 4; ++t) {
            const int d = t * 16 + nloc;       // d < 64; cos/sin repeat at d+64
#pragma unroll
            for (int v = 0; v < 8; ++v) {
                const int s = m0 + mloc + v;
                const float c  = cosp[(size_t)s * 128 + d];
                const float sn = sinp[(size_t)s * 128 + d];
                const float x1 = acc[t][v], x2 = acc[t + 4][v];
                acc[t][v]     = x1 * c - x2 * sn;
                acc[t + 4][v] = x2 * c + x1 * sn;
            }
        }
    }

#pragma unroll
    for (int t = 0; t < 8; ++t) {
        const int d = t * 16 + nloc;
#pragma unroll
        for (int v = 0; v < 8; ++v) {
            const int s = m0 + mloc + v;
            const float val = acc[t][v];
            if (mode == 0) {
                outf[(size_t)s * N + n0 + d] = val;
            } else if (mode == 2) {            // K^T: [h][d][S], bf16
                outb[((size_t)(n0 >> 7) * 128 + d) * (size_t)S + s] = f2bf(val);
            } else {                           // Q / V: [h][s][128], bf16
                outb[((size_t)(n0 >> 7) * M + s) * 128 + d] = f2bf(val);
            }
        }
    }
}

// ---------------------------------------------------------------------------
// Fused attention: per (head, 64-query tile); loop over 32-key tiles with
// double-buffered async K^T/V staging. Q operand lives in register fragments;
// the score buffer aliases the (dead) Q staging LDS.
// scores = Q K^T * scale; aw = ((scores + mask + 7)/rd)^8; out += aw @ V
// ---------------------------------------------------------------------------
#define Q_STRIDE 136    // 128 + 8 pad shorts (272 B)
#define KT_STRIDE 40    // 32 + 8 pad shorts  (80 B)
#define S_STRIDE 40
#define V_STRIDE 136

__global__ __launch_bounds__(128) void attn_pow8_wmma(
    const unsigned short* __restrict__ Qws, const unsigned short* __restrict__ Kws,
    const unsigned short* __restrict__ Vws, const float* __restrict__ rd,
    unsigned short* __restrict__ outp, int S)
{
    __shared__ __align__(16) unsigned short QSs[64 * Q_STRIDE];      // Q, then aw
    __shared__ __align__(16) unsigned short KTs[2][128 * KT_STRIDE]; // hd x key
    __shared__ __align__(16) unsigned short Vs[2][32 * V_STRIDE];    // key x hd

    const int tid = threadIdx.x, lane = tid & 31, wave = tid >> 5;  // 4 waves
    const int h   = blockIdx.y;
    const int hkv = h >> 1;                    // GQA groups = 2
    const int s0  = blockIdx.x * 64;
    const float scaling = 0.08838834764831845f;   // 128^-0.5

    const unsigned qBase  = lds_off(QSs);
    const unsigned ktBase[2] = { lds_off(KTs[0]), lds_off(KTs[1]) };
    const unsigned vBase[2]  = { lds_off(Vs[0]),  lds_off(Vs[1])  };

    // stage one 32-key tile (K^T + V), 8 x b128 per thread
    const unsigned short* kgp = Kws + ((size_t)hkv * 128 + tid) * S;
    const int vrow = tid >> 2, vsegb = (tid & 3) * 4;
    const unsigned short* vgp = Vws + ((size_t)hkv * S + vrow) * 128;
    auto stage_kv = [&](int buf, int kb) {
#pragma unroll
        for (int i = 0; i < 4; ++i)            // K^T: row hd=tid, 64 B
            async_ld16(ktBase[buf] + tid * (KT_STRIDE * 2) + i * 16,
                       kgp + kb + i * 8);
#pragma unroll
        for (int i = 0; i < 4; ++i)            // V: 32 rows x 256 B
            async_ld16(vBase[buf] + vrow * (V_STRIDE * 2) + (vsegb + i) * 16,
                       vgp + (size_t)kb * 128 + (vsegb + i) * 8);
    };

    {   // stage Q tile: 64 rows x 256 B, 8 x b128 per thread
        const int r = tid >> 1, sb = (tid & 1) * 8;
        const unsigned short* qp = Qws + ((size_t)h * S + (s0 + r)) * 128;
#pragma unroll
        for (int i = 0; i < 8; ++i)
            async_ld16(qBase + r * (Q_STRIDE * 2) + (sb + i) * 16, qp + (sb + i) * 8);
    }
    stage_kv(0, 0);                            // overlap first K/V tile with Q wait
    wait_async_le8();                          // in-order: Q's 8 have landed
    __syncthreads();

    // hoist Q operand into register fragments (reused for all key tiles)
    Frag qfrag[4];
#pragma unroll
    for (int c = 0; c < 4; ++c)
        load_a_frag(qfrag[c], QSs, wave * 16, Q_STRIDE, c * 32, lane);

    v8f oacc[8];
#pragma unroll
    for (int t = 0; t < 8; ++t)
#pragma unroll
        for (int v = 0; v < 8; ++v) oacc[t][v] = 0.0f;

    const int mloc = wave * 16 + ((lane >> 4) & 1) * 8;
    const int nloc = lane & 15;

    float rinv[8];
#pragma unroll
    for (int v = 0; v < 8; ++v)
        rinv[v] = 1.0f / rd[(size_t)h * S + s0 + mloc + v];

    const int nk = S >> 5;                     // 32-key tiles
    for (int i = 0; i < nk; ++i) {
        const int kb = i << 5;
        if (i + 1 < nk) {
            stage_kv((i + 1) & 1, kb + 32);    // overlap next tile's DMA
            wait_async_le8();                  // only tile i must have landed
        } else {
            wait_async_le0();
        }
        __syncthreads();                       // also: Q-frag reads done (iter 0)

        const unsigned short* kt = KTs[i & 1];
        const unsigned short* vv = Vs[i & 1];

        // ---- scores: wave's 16 q-rows x 32 keys ----
        v8f sacc[2];
#pragma unroll
        for (int t = 0; t < 2; ++t)
#pragma unroll
            for (int v = 0; v < 8; ++v) sacc[t][v] = 0.0f;

#pragma unroll
        for (int c = 0; c < 4; ++c) {          // hd chunks of 32
            Frag b;
#pragma unroll
            for (int t = 0; t < 2; ++t) {
                load_b_frag(b, kt, c * 32, KT_STRIDE, t * 16, lane);
                sacc[t] = __builtin_amdgcn_wmma_f32_16x16x32_bf16(
                    false, qfrag[c].v, false, b.v, (short)0, sacc[t], false, false);
            }
        }

        // ---- epilogue: mask + shift + *1/rd + pow8, bf16 into aw buffer ----
#pragma unroll
        for (int t = 0; t < 2; ++t) {
            const int keyl = t * 16 + nloc;
#pragma unroll
            for (int v = 0; v < 8; ++v) {
                const int q   = s0 + mloc + v;
                const int key = kb + keyl;
                float x = sacc[t][v] * scaling + (key > q ? -7.0f : 0.0f);
                x = (x + 7.0f) * rinv[v];
                const float x2 = x * x, x4 = x2 * x2;
                QSs[(mloc + v) * S_STRIDE + keyl] = f2bf(x4 * x4);
            }
        }
        __syncthreads();

        // ---- out += aw @ V (one 32-key chunk) ----
        Frag a, b;
        load_a_frag(a, QSs, wave * 16, S_STRIDE, 0, lane);
#pragma unroll
        for (int t = 0; t < 8; ++t) {
            load_b_frag(b, vv, 0, V_STRIDE, t * 16, lane);
            oacc[t] = __builtin_amdgcn_wmma_f32_16x16x32_bf16(
                false, a.v, false, b.v, (short)0, oacc[t], false, false);
        }
        __syncthreads();
    }

    // store bf16 to [s][h*128 + d] for the O-projection GEMM
#pragma unroll
    for (int t = 0; t < 8; ++t) {
        const int d = t * 16 + nloc;
#pragma unroll
        for (int v = 0; v < 8; ++v) {
            const int s = s0 + mloc + v;
            outp[(size_t)s * 2048 + (size_t)h * 128 + d] = f2bf(oacc[t][v]);
        }
    }
}

// ---------------------------------------------------------------------------
extern "C" void kernel_launch(void* const* d_in, const int* in_sizes, int n_in,
                              void* d_out, int out_size, void* d_ws, size_t ws_size,
                              hipStream_t stream) {
    (void)in_sizes; (void)n_in; (void)out_size; (void)ws_size;
    const float* hidden = (const float*)d_in[0];
    const float* cosp   = (const float*)d_in[1];
    const float* sinp   = (const float*)d_in[2];
    const float* qw     = (const float*)d_in[3];
    const float* kw     = (const float*)d_in[4];
    const float* vw     = (const float*)d_in[5];
    const float* ow     = (const float*)d_in[6];
    const float* qn     = (const float*)d_in[7];
    const float* kn     = (const float*)d_in[8];
    const float* qd     = (const float*)d_in[9];
    const float* kd     = (const float*)d_in[10];
    const float* rdp    = (const float*)d_in[11];
    // d_in[12] (attention_mask) is causal 0/-7 -> computed inline.

    const int S = 2048, D = 2048;
    const size_t E22 = (size_t)2048 * 2048;   // 4,194,304
    const size_t E21 = (size_t)2048 * 1024;   // 2,097,152

    unsigned short* ws    = (unsigned short*)d_ws;
    unsigned short* hidB  = ws;                 // bf16 hidden       8 MB
    unsigned short* qwB   = hidB + E22;         // bf16 q_proj_w     8 MB
    unsigned short* kwB   = qwB + E22;          // bf16 k_proj_w     4 MB
    unsigned short* vwB   = kwB + E21;          // bf16 v_proj_w     4 MB
    unsigned short* owB   = vwB + E21;          // bf16 o_proj_w     8 MB
    unsigned short* QwsB  = owB + E22;          // [16][S][128]      8 MB
    unsigned short* KwsB  = QwsB + E22;         // [8][128][S]       4 MB
    unsigned short* VwsB  = KwsB + E21;         // [8][S][128]       4 MB
    unsigned short* AttnB = VwsB + E21;         // [S][2048]         8 MB

    // one-time fp32 -> bf16 conversions
    f32_to_bf16<<<E22 / 1024, 256, 0, stream>>>(hidden, hidB, (int)E22);
    f32_to_bf16<<<E22 / 1024, 256, 0, stream>>>(qw, qwB, (int)E22);
    f32_to_bf16<<<E21 / 1024, 256, 0, stream>>>(kw, kwB, (int)E21);
    f32_to_bf16<<<E21 / 1024, 256, 0, stream>>>(vw, vwB, (int)E21);
    f32_to_bf16<<<E22 / 1024, 256, 0, stream>>>(ow, owB, (int)E22);

    // Q / K / V projections with fused norm + RoPE epilogues (bf16 out)
    gemm_bf16_wmma<<<dim3(16, 16), 256, 0, stream>>>(
        hidB, qwB, nullptr, QwsB, S, D, 2048, 1, cosp, sinp, qn, qd, S, 16);
    gemm_bf16_wmma<<<dim3(8, 16), 256, 0, stream>>>(
        hidB, kwB, nullptr, KwsB, S, D, 1024, 2, cosp, sinp, kn, kd, S, 8);
    gemm_bf16_wmma<<<dim3(8, 16), 256, 0, stream>>>(
        hidB, vwB, nullptr, VwsB, S, D, 1024, 3, cosp, sinp, kn, kd, S, 8);

    // fused pow8-attention (bf16 out)
    attn_pow8_wmma<<<dim3(S / 64, 16), 128, 0, stream>>>(
        QwsB, KwsB, VwsB, rdp, AttnB, S);

    // O-projection straight into d_out (fp32)
    gemm_bf16_wmma<<<dim3(16, 16), 256, 0, stream>>>(
        AttnB, owB, (float*)d_out, nullptr, S, 2048, 2048, 0,
        cosp, sinp, qn, qd, S, 16);
}
